// SignatureRNN_24996709662820
// MI455X (gfx1250) — compile-verified, matrix operand order checked by reading, not run on hardware
//
#include <hip/hip_runtime.h>
#include <hip/hip_bf16.h>

typedef __bf16 bf16_t;
typedef __attribute__((ext_vector_type(16))) __bf16 v16bf;
typedef __attribute__((ext_vector_type(8)))  __bf16 v8bf;
typedef __attribute__((ext_vector_type(8)))  float  v8f;

// Problem constants
#define BB   16
#define SS   2048
#define DIN  8
#define HH   512
#define NG4  2048          // 4*H
#define KA   832           // padded 827 -> 832
#define MROWS (BB*SS)      // 32768
#define NWG  64            // workgroups in recurrent kernel
#define JW   (HH/NWG)      // h indices per WG = 8

// ---------------------------------------------------------------------------
// WMMA fragment loader: 16-bit A/B matrix, row-major source [row][k], ld elems.
// Lane l: row = row0 + (l&15), holds K = koff..koff+7 and koff+16..koff+23
// where koff = (l>>4)*8 (per CDNA5 ISA 16-bit 16x32 layout).
// ---------------------------------------------------------------------------
__device__ __forceinline__ v16bf load_frag(const bf16_t* __restrict__ base,
                                           int ld, int row0, int kbase, int lane) {
  int r    = lane & 15;
  int koff = (lane >> 4) << 3;
  const bf16_t* p = base + (size_t)(row0 + r) * ld + kbase + koff;
  v8bf lo = *(const v8bf*)(p);
  v8bf hi = *(const v8bf*)(p + 16);
  v16bf out;
#pragma unroll
  for (int i = 0; i < 8; ++i) { out[i] = lo[i]; out[i + 8] = hi[i]; }
  return out;
}

__device__ __forceinline__ float sigmoidf_(float x) {
  return 1.0f / (1.0f + __expf(-x));
}

// ---------------------------------------------------------------------------
// Kernel A: windowed signatures -> bf16 x_aug [32768 rows (t*16+b)][832]
// One 9-lane group per position (3 positions per wave). Lane i owns dim-row i
// of s1/s2/s3, so Chen's update is fully lane-local (no cross-lane traffic).
// ---------------------------------------------------------------------------
__global__ __launch_bounds__(256) void sig_kernel(const float* __restrict__ x,
                                                  bf16_t* __restrict__ Xa) {
  int lane = threadIdx.x & 31;
  int wid  = threadIdx.x >> 5;
  int sub  = lane / 9;            // 0..2 active, 3 -> idle (lanes 27..31)
  int i    = lane - sub * 9;      // 0..8
  long pos = (long)blockIdx.x * 24 + wid * 3 + sub;
  if (sub >= 3 || pos >= MROWS) return;
  int t = (int)(pos >> 4);
  int b = (int)(pos & 15);

  float s1 = 0.f;
  float s2[9];  float s3[81];
#pragma unroll
  for (int j = 0; j < 9; ++j) s2[j] = 0.f;
#pragma unroll
  for (int z = 0; z < 81; ++z) s3[z] = 0.f;

  float prev[9], dx[9];
  const float* xb = x + (size_t)b * SS * DIN;

#pragma unroll 1
  for (int w = 0; w < 10; ++w) {
    int sidx = t - 9 + w;
    sidx = sidx < 0 ? 0 : sidx;          // upper clip unnecessary (<= t)
    float cur[9];
#pragma unroll
    for (int d = 0; d < 8; ++d) cur[d] = xb[sidx * 8 + d];
    cur[8] = (float)sidx * (1.0f / 2047.0f);
#pragma unroll
    for (int d = 0; d < 9; ++d) {
      dx[d] = (w == 0) ? cur[d] : (cur[d] - prev[d]);
      prev[d] = cur[d];
    }
    float dxi = dx[i];
    // s3[j][k] += (s2[j] + 0.5*s1*dx[j] + dx[i]*dx[j]/6) * dx[k]   (old s1,s2)
#pragma unroll
    for (int j = 0; j < 9; ++j) {
      float coef = s2[j] + 0.5f * s1 * dx[j] + dxi * dx[j] * (1.0f / 6.0f);
#pragma unroll
      for (int k = 0; k < 9; ++k) s3[j * 9 + k] += coef * dx[k];
    }
#pragma unroll
    for (int j = 0; j < 9; ++j) s2[j] += (s1 + 0.5f * dxi) * dx[j];
    s1 += dxi;
  }

  bf16_t* row = Xa + (size_t)pos * KA;
  if (i < 8) row[i] = (bf16_t)xb[t * 8 + i];          // raw x part
  if (i == 0) {
#pragma unroll
    for (int p2 = 827; p2 < 832; ++p2) row[p2] = (bf16_t)0.f;  // K pad
  }
  row[8 + i] = (bf16_t)s1;
#pragma unroll
  for (int j = 0; j < 9; ++j) row[17 + i * 9 + j] = (bf16_t)s2[j];
#pragma unroll
  for (int j = 0; j < 9; ++j)
#pragma unroll
    for (int k = 0; k < 9; ++k)
      row[98 + i * 81 + j * 9 + k] = (bf16_t)s3[j * 9 + k];
}

// ---------------------------------------------------------------------------
// Kernel B: W_ih fp32 [2048][827] -> bf16 [2048][832] (zero-padded K)
// ---------------------------------------------------------------------------
__global__ __launch_bounds__(256) void cvt_wih(const float* __restrict__ W,
                                               bf16_t* __restrict__ Wb) {
  int idx = blockIdx.x * 256 + threadIdx.x;
  if (idx >= NG4 * KA) return;
  int n = idx / KA, k = idx - n * KA;
  Wb[idx] = (k < 827) ? (bf16_t)W[(size_t)n * 827 + k] : (bf16_t)0.f;
}

// ---------------------------------------------------------------------------
// Kernel C: zero h double-buffer (bf16) and the global barrier counter
// ---------------------------------------------------------------------------
__global__ __launch_bounds__(256) void init_state(bf16_t* __restrict__ hbuf,
                                                  unsigned* __restrict__ ctr) {
  int idx = blockIdx.x * 256 + threadIdx.x;
  if (idx < 2 * BB * HH) hbuf[idx] = (bf16_t)0.f;
  if (idx == 0) *ctr = 0u;
}

// ---------------------------------------------------------------------------
// Kernel D: pre-gates GEMM. C[32768x2048] = Xa[32768x832] * Wb^T + bias,
// stored bf16. 8 waves/WG (2x4), each wave a 32x64 C region (2x4 wmma tiles):
// per k-step 8 WMMA vs 6 fragment loads (~21 flop/byte from cache).
// ---------------------------------------------------------------------------
__global__ __launch_bounds__(256) void gemm_pregates(const bf16_t* __restrict__ Xa,
                                                     const bf16_t* __restrict__ Wb,
                                                     const float* __restrict__ bih,
                                                     const float* __restrict__ bhh,
                                                     bf16_t* __restrict__ Gpre) {
  int wid  = threadIdx.x >> 5;
  int lane = threadIdx.x & 31;
  int m0 = blockIdx.x * 64  + (wid & 1) * 32;   // 512 blocks in M
  int n0 = blockIdx.y * 256 + (wid >> 1) * 64;  // 8 blocks in N

  v8f acc[2][4];
#pragma unroll
  for (int mi = 0; mi < 2; ++mi)
#pragma unroll
    for (int ni = 0; ni < 4; ++ni) acc[mi][ni] = (v8f){};

  for (int kb = 0; kb < KA; kb += 32) {
    v16bf a[2], bfr[4];
#pragma unroll
    for (int mi = 0; mi < 2; ++mi) a[mi] = load_frag(Xa, KA, m0 + 16 * mi, kb, lane);
#pragma unroll
    for (int ni = 0; ni < 4; ++ni) bfr[ni] = load_frag(Wb, KA, n0 + 16 * ni, kb, lane);
#pragma unroll
    for (int mi = 0; mi < 2; ++mi)
#pragma unroll
      for (int ni = 0; ni < 4; ++ni)
        acc[mi][ni] = __builtin_amdgcn_wmma_f32_16x16x32_bf16(
            false, a[mi], false, bfr[ni], (short)0, acc[mi][ni], false, false);
  }

  int cl = lane & 15, half = lane >> 4;
#pragma unroll
  for (int ni = 0; ni < 4; ++ni) {
    int n = n0 + ni * 16 + cl;
    float bias = bih[n] + bhh[n];
#pragma unroll
    for (int mi = 0; mi < 2; ++mi) {
#pragma unroll
      for (int r = 0; r < 8; ++r) {
        int m = m0 + mi * 16 + half * 8 + r;
        Gpre[(size_t)m * NG4 + n] = (bf16_t)(acc[mi][ni][r] + bias);
      }
    }
  }
}

// ---------------------------------------------------------------------------
// Kernel E: persistent LSTM recurrence. 64 WGs x 64 threads (2 waves).
// WG g owns h indices j in [g*8, g*8+8) => gate columns {gate*512+g*8+jl}.
// Its 32x512 W_hh slice lives in LDS (bf16, ds_load_b128 feeds WMMA B).
// Per step: pre-gate loads issued first (overlap WMMA chain), 2 waves x 16
// WMMA over K=512, gate nonlinearity, double-buffered bf16 h in global +
// device-wide atomic-counter barrier; prefetch next step's pre-gates.
// ---------------------------------------------------------------------------
__global__ __launch_bounds__(64) void lstm_rec(const bf16_t* __restrict__ Gpre,
                                               const float* __restrict__ Whh,
                                               bf16_t* __restrict__ hbuf,
                                               unsigned* __restrict__ ctr,
                                               float* __restrict__ out) {
  __shared__ __align__(16) bf16_t sW[32][HH];     // 32 KB
  __shared__ float sGate[4][BB][JW];              // 2 KB
  __shared__ float sC[BB][JW];                    // 0.5 KB

  const int g    = blockIdx.x;
  const int tid  = threadIdx.x;
  const int lane = tid & 31;
  const int wv   = tid >> 5;
  const int cl   = lane & 15, half = lane >> 4;

  // Preload + convert W_hh slice: n_local = gate*8 + jl  ->  n = gate*512+g*8+jl
  for (int idx = tid; idx < 32 * HH; idx += 64) {
    int nl = idx >> 9, k = idx & (HH - 1);
    int gate = nl >> 3, jl = nl & 7;
    sW[nl][k] = (bf16_t)Whh[((size_t)(gate * HH + g * JW + jl)) * HH + k];
  }
  for (int idx = tid; idx < BB * JW; idx += 64) sC[idx >> 3][idx & 7] = 0.f;
  __syncthreads();

  float* ys = out;
  float* hn = out + (size_t)BB * SS * HH;
  float* cn = hn + BB * HH;

  // column mapping for this thread's C tile
  const int nl   = 16 * wv + cl;          // n_local 0..31
  const int gate = nl >> 3, jl = nl & 7;
  const int ncol = gate * HH + g * JW + jl;

  for (int t = 0; t < SS; ++t) {
    const bf16_t* hb = hbuf + (size_t)(t & 1) * (BB * HH);

    // issue this step's pre-gate loads first so they overlap the WMMA chain
    const bf16_t* gp = Gpre + ((size_t)t * BB) * NG4 + ncol;
    float gpre_r[8];
#pragma unroll
    for (int r = 0; r < 8; ++r)
      gpre_r[r] = (float)gp[(size_t)(half * 8 + r) * NG4];
    if (t + 1 < SS)
      __builtin_prefetch(Gpre + ((size_t)(t + 1) * BB) * NG4 + ncol, 0, 1);

    v8f acc = {};
    for (int kb = 0; kb < HH; kb += 32) {
      v16bf a = load_frag(hb, HH, 0, kb, lane);   // rows = batch 0..15
      v16bf bm;
      {
        const bf16_t* p = &sW[nl][kb + (half << 3)];
        v8bf lo = *(const v8bf*)(p);
        v8bf hi = *(const v8bf*)(p + 16);
#pragma unroll
        for (int q = 0; q < 8; ++q) { bm[q] = lo[q]; bm[q + 8] = hi[q]; }
      }
      acc = __builtin_amdgcn_wmma_f32_16x16x32_bf16(false, a, false, bm, (short)0, acc, false, false);
    }

    // add pre-gates, scatter to sGate[gate][b][jl]
#pragma unroll
    for (int r = 0; r < 8; ++r) {
      int bb = half * 8 + r;
      sGate[gate][bb][jl] = acc[r] + gpre_r[r];
    }
    __syncthreads();

    // gate nonlinearity + state update (128 elements over 64 threads)
    for (int e = tid; e < BB * JW; e += 64) {
      int bb = e >> 3, jj = e & 7;
      float iv = sGate[0][bb][jj], fv = sGate[1][bb][jj];
      float gv = sGate[2][bb][jj], ov = sGate[3][bb][jj];
      float c = sigmoidf_(fv) * sC[bb][jj] + sigmoidf_(iv) * tanhf(gv);
      float h = sigmoidf_(ov) * tanhf(c);
      sC[bb][jj] = c;
      int j = g * JW + jj;
      ys[((size_t)bb * SS + t) * HH + j] = h;
      hbuf[(size_t)((t + 1) & 1) * (BB * HH) + (size_t)bb * HH + j] = (bf16_t)h;
      if (t == SS - 1) { hn[bb * HH + j] = h; cn[bb * HH + j] = c; }
    }
    __threadfence();        // release: make h writes visible device-wide
    __syncthreads();

    if (tid == 0) {
      atomicAdd(ctr, 1u);
      unsigned target = (unsigned)(t + 1) * NWG;
      while (atomicAdd(ctr, 0u) < target) { __builtin_amdgcn_s_sleep(2); }
    }
    __syncthreads();
    __threadfence();        // acquire: invalidate before reading peers' h
  }
}

// ---------------------------------------------------------------------------
// Launcher
// ---------------------------------------------------------------------------
extern "C" void kernel_launch(void* const* d_in, const int* in_sizes, int n_in,
                              void* d_out, int out_size, void* d_ws, size_t ws_size,
                              hipStream_t stream) {
  (void)in_sizes; (void)n_in; (void)out_size;
  const float* x   = (const float*)d_in[0];
  const float* Wih = (const float*)d_in[1];
  const float* Whh = (const float*)d_in[2];
  const float* bih = (const float*)d_in[3];
  const float* bhh = (const float*)d_in[4];

  const size_t OFF_XA   = 0;
  const size_t SZ_XA    = (size_t)MROWS * KA * sizeof(bf16_t);        // 54,525,952
  const size_t OFF_WB   = OFF_XA + SZ_XA;
  const size_t SZ_WB    = (size_t)NG4 * KA * sizeof(bf16_t);          //  3,407,872
  const size_t OFF_GP   = OFF_WB + SZ_WB;
  const size_t SZ_GP    = (size_t)MROWS * NG4 * sizeof(bf16_t);       // 134,217,728
  const size_t OFF_HB   = OFF_GP + SZ_GP;
  const size_t SZ_HB    = (size_t)2 * BB * HH * sizeof(bf16_t);
  const size_t OFF_CTR  = OFF_HB + ((SZ_HB + 255) & ~(size_t)255);
  const size_t NEEDED   = OFF_CTR + 256;
  if (ws_size < NEEDED) return;

  char* ws = (char*)d_ws;
  bf16_t*   Xa   = (bf16_t*)(ws + OFF_XA);
  bf16_t*   Wb   = (bf16_t*)(ws + OFF_WB);
  bf16_t*   Gpre = (bf16_t*)(ws + OFF_GP);
  bf16_t*   hbuf = (bf16_t*)(ws + OFF_HB);
  unsigned* ctr  = (unsigned*)(ws + OFF_CTR);
  float*    out  = (float*)d_out;

  sig_kernel<<<dim3((MROWS + 23) / 24), dim3(256), 0, stream>>>(x, Xa);
  cvt_wih<<<dim3((NG4 * KA + 255) / 256), dim3(256), 0, stream>>>(Wih, Wb);
  init_state<<<dim3(64), dim3(256), 0, stream>>>(hbuf, ctr);
  gemm_pregates<<<dim3(MROWS / 64, NG4 / 256), dim3(256), 0, stream>>>(Xa, Wb, bih, bhh, Gpre);
  lstm_rec<<<dim3(NWG), dim3(64), 0, stream>>>(Gpre, Whh, hbuf, ctr, out);
}
